// SelfAttention_1108101562437
// MI455X (gfx1250) — matrix operand order repaired and checked by reading, not compiled, not run on hardware
//
#include <hip/hip_runtime.h>
#include <hip/hip_bf16.h>

#define HID   1024
#define HEADS 16
#define HDIM  64
#define BB    4
#define SS    2048
#define MTOK  (BB * SS)   // 8192 token rows

typedef __attribute__((ext_vector_type(16))) __bf16 bf16x16;
typedef __attribute__((ext_vector_type(8)))  __bf16 bf16x8;
typedef __attribute__((ext_vector_type(4)))  __bf16 bf16x4;
typedef __attribute__((ext_vector_type(8)))  float  f32x8;

union BF16x16 { bf16x16 v; bf16x8 h[2]; unsigned u[8]; };

__device__ __forceinline__ f32x8 wmma_bf16(bf16x16 a, bf16x16 b, f32x8 c) {
    // D(f32 16x16) = A(bf16 16x32) * B(bf16 32x16) + C
    return __builtin_amdgcn_wmma_f32_16x16x32_bf16(false, a, false, b,
                                                   (short)0, c, false, false);
}

// lane i <-> lane i^16 exchange, pure VALU (v_permlanex16_b32), no LDS
__device__ __forceinline__ unsigned xor16_u32(unsigned x) {
    return __builtin_amdgcn_permlanex16(x, x, 0x76543210u, 0xfedcba98u,
                                        false, false);
}
__device__ __forceinline__ float xor16_f32(float x) {
    union { float f; unsigned u; } a, r;
    a.f = x;
    r.u = xor16_u32(a.u);
    return r.f;
}
__device__ __forceinline__ unsigned pack_bf16(float a, float b) {
    union { __bf16 h[2]; unsigned u; } x;
    x.h[0] = (__bf16)a; x.h[1] = (__bf16)b;
    return x.u;
}
// raw v_exp_f32 (base-2, flush-to-zero below 2^-126): exactly right for
// softmax probabilities after max subtraction (args are always <= 0).
__device__ __forceinline__ float exp2_raw(float x) {
    return __builtin_amdgcn_exp2f(x);
}

// ---------------------------------------------------------------------------
// fp32 -> bf16 conversion, 4 elements per thread
// ---------------------------------------------------------------------------
__global__ void cvt_f32_bf16(const float* __restrict__ in,
                             __bf16* __restrict__ out, int n4) {
    int i = blockIdx.x * blockDim.x + threadIdx.x;
    if (i < n4) {
        float4 f = ((const float4*)in)[i];
        bf16x4 o;
        o.x = (__bf16)f.x; o.y = (__bf16)f.y;
        o.z = (__bf16)f.z; o.w = (__bf16)f.w;
        *(bf16x4*)(out + 4 * (size_t)i) = o;
    }
}

// ---------------------------------------------------------------------------
// GEMM: Y[m,n] = (sum_k A[m,k] * W[n,k] + bias[n]) * scale
// Workgroup = 4 waves = 64x64 tile; each wave: 16 rows x 64 cols.
// W tiles (64 rows x 32 k, 4 KB) are shared by all 4 waves -> staged through
// LDS with double-buffered global_load_async_to_lds_b128 (ASYNCcnt-tracked).
// MODE 0: store bf16 to [B,H,S,D]   (Q with softmax scale folded in, K)
// MODE 1: store bf16 to [B,H,D,S]   (V, pre-transposed for attention)
// MODE 2: store f32  to [M, HID]    (final output projection)
// ---------------------------------------------------------------------------
template<int MODE>
__global__ __launch_bounds__(128)
void gemm_bf16(const __bf16* __restrict__ A,
               const __bf16* __restrict__ W,
               const float*  __restrict__ bias,
               void* __restrict__ out, float scale)
{
    __shared__ __bf16 wtile[2][64 * 32];   // 2 x 4 KB double buffer

    const int tid   = threadIdx.x;
    const int lane  = tid & 31;
    const int wave  = tid >> 5;
    const int half  = lane >> 4;
    const int l16   = lane & 15;
    const int mtile = blockIdx.x * 64 + wave * 16;
    const int ntile = blockIdx.y * 64;

    // async-copy mapping: thread copies 2 x 16B chunks of the 4 KB tile.
    // tile row stride = 64 B (32 bf16). chunk byte c: row=c/64, inrow=c%64.
    const int c0 = tid * 16;
    const int r0 = c0 >> 6, b0 = c0 & 63;
    const int c1 = c0 + 2048;
    const int r1 = c1 >> 6, b1 = c1 & 63;
    const char* Wb = (const char*)(W + (size_t)ntile * HID);

    auto issue = [&](int buf, int k0) {
        // generic LDS pointer truncates to the 32-bit LDS offset (ISA 10.2)
        unsigned lb = (unsigned)(uintptr_t)(&wtile[buf][0]);
        const char* g0 = Wb + (size_t)r0 * (HID * 2) + k0 * 2 + b0;
        const char* g1 = Wb + (size_t)r1 * (HID * 2) + k0 * 2 + b1;
        unsigned l0 = lb + (unsigned)c0, l1 = lb + (unsigned)c1;
        asm volatile("global_load_async_to_lds_b128 %0, %1, off"
                     :: "v"(l0), "v"(g0) : "memory");
        asm volatile("global_load_async_to_lds_b128 %0, %1, off"
                     :: "v"(l1), "v"(g1) : "memory");
    };

    // A fragment base: lane holds row (mtile+l16), K elems {16h+j}
    const __bf16* arow = A + (size_t)(mtile + l16) * HID + half * 16;

    f32x8 acc[4] = {};
    const int NSTEP = HID / 32;
    issue(0, 0);

    for (int i = 0; i < NSTEP; ++i) {
        const int k0 = i * 32;
        if (i + 1 < NSTEP) {
            issue((i + 1) & 1, k0 + 32);
            asm volatile("s_wait_asynccnt 0x2" ::: "memory"); // current tile done
        } else {
            asm volatile("s_wait_asynccnt 0x0" ::: "memory");
        }
        __syncthreads();

        __builtin_prefetch(arow + k0 + 128, 0, 1);   // global_prefetch_b8
        BF16x16 af;
        af.h[0] = *(const bf16x8*)(arow + k0);        // K = k0 + 16h + j
        af.h[1] = *(const bf16x8*)(arow + k0 + 8);    // K = k0 + 16h + 8 + j

        const __bf16* wl = &wtile[i & 1][0];
#pragma unroll
        for (int cb = 0; cb < 4; ++cb) {
            // B[k][n] = W[n][k]: lane(n=l16, khalf=half), 16 contiguous bf16
            bf16x16 bfrag = *(const bf16x16*)(wl + (cb * 16 + l16) * 32
                                                 + half * 16);
            acc[cb] = wmma_bf16(af.v, bfrag, acc[cb]);
        }
        __syncthreads();
    }

    // C layout: lane holds (m = v + 8*half, n = l16) for VGPR v
#pragma unroll
    for (int cb = 0; cb < 4; ++cb) {
        const int n   = ntile + cb * 16 + l16;
        const float bv = bias[n];
#pragma unroll
        for (int v = 0; v < 8; ++v) {
            const int m  = mtile + v + 8 * half;
            const float y = (acc[cb][v] + bv) * scale;
            if (MODE == 0) {                     // [B,H,S,D] bf16
                const int b = m / SS, s = m % SS;
                const int h = n / HDIM, d = n % HDIM;
                ((__bf16*)out)[(((size_t)(b * HEADS + h)) * SS + s) * HDIM + d]
                    = (__bf16)y;
            } else if (MODE == 1) {              // [B,H,D,S] bf16
                const int b = m / SS, s = m % SS;
                const int h = n / HDIM, d = n % HDIM;
                ((__bf16*)out)[(((size_t)(b * HEADS + h)) * HDIM + d) * SS + s]
                    = (__bf16)y;
            } else {                             // [M,HID] f32
                ((float*)out)[(size_t)m * HID + n] = y;
            }
        }
    }
}

// ---------------------------------------------------------------------------
// Flash attention, one (b,h) per blockIdx.y, 4 waves x 16 query rows per WG.
// Scores computed TRANSPOSED (A=K, B=Q): C-layout = (kk = v+8h, q = lane&15).
// Context also computed TRANSPOSED (A=V^T, B=P^T): C-layout = (d, q=lane&15),
// so all softmax state (m, s, corr) is consumed by the OWN lane — the only
// cross-lane traffic is v_permlanex16 (no LDS at all in this kernel).
// Q arrives pre-scaled by log2(e)/sqrt(D); softmax runs in base 2 with raw
// v_exp_f32.  64 keys per iteration: 16 WMMAs per softmax round.
// ---------------------------------------------------------------------------
__global__ __launch_bounds__(128)
void flash_attn(const __bf16* __restrict__ Qp,   // [B*H, S, D] (pre-scaled)
                const __bf16* __restrict__ Kp,   // [B*H, S, D]
                const __bf16* __restrict__ Vt,   // [B*H, D, S]
                __bf16* __restrict__ Oa)         // [B, S, H*D]
{
    const int lane  = threadIdx.x & 31;
    const int wave  = threadIdx.x >> 5;
    const int half  = lane >> 4;
    const int l16   = lane & 15;
    const int bh    = blockIdx.y;
    const int qbase = blockIdx.x * 64 + wave * 16;

    const __bf16* Qb = Qp + (size_t)bh * SS * HDIM;
    const __bf16* Kb = Kp + (size_t)bh * SS * HDIM;
    const __bf16* Vb = Vt + (size_t)bh * HDIM * SS;

    // Q as B-matrix: lane(q=l16, dhalf=half) reads 16 contiguous bf16 of row q
    bf16x16 qf[2];
#pragma unroll
    for (int db = 0; db < 2; ++db)
        qf[db] = *(const bf16x16*)(Qb + (size_t)(qbase + l16) * HDIM
                                      + db * 32 + half * 16);

    f32x8 acc[4] = {};                 // context^T, d-blocks 0..3; q = l16
    float m_run = -3.0e38f, s_run = 0.f;   // log2-domain, row q = l16

    for (int k0 = 0; k0 < SS; k0 += 64) {
        // --- four transposed score tiles Ct[kk,q], kk blocks of 16 ---
        f32x8 sc[4];
#pragma unroll
        for (int t = 0; t < 4; ++t) {
            const __bf16* krow = Kb + (size_t)(k0 + t * 16 + l16) * HDIM
                                    + half * 16;
            f32x8 c = {};
#pragma unroll
            for (int db = 0; db < 2; ++db) {
                BF16x16 kf;                       // K rows as A-matrix
                kf.h[0] = *(const bf16x8*)(krow + db * 32);
                kf.h[1] = *(const bf16x8*)(krow + db * 32 + 8);
                c = wmma_bf16(kf.v, qf[db], c);
            }
            sc[t] = c;
        }
        // prefetch next key/value block while softmax runs
        __builtin_prefetch(Kb + (size_t)(k0 + 64 + l16) * HDIM, 0, 1);
        __builtin_prefetch(Vb + (size_t)l16 * SS + k0 + 64, 0, 1);

        // --- online softmax (base-2) for row q = l16 over 64 keys ---
        float lmax = -3.0e38f;
#pragma unroll
        for (int t = 0; t < 4; ++t)
#pragma unroll
            for (int v = 0; v < 8; ++v) lmax = fmaxf(lmax, sc[t][v]);
        lmax = fmaxf(lmax, xor16_f32(lmax));            // join kk halves
        const float m_new = fmaxf(m_run, lmax);
        const float corr  = exp2_raw(m_run - m_new);

        // build P^T as two WMMA B-fragments (kk 0..31 and 32..63).
        // B VGPR i holds elements jj=2i,2i+1 with kk = 16*half + jj, sourced
        // from score tile t = 2g+half; jj<8 -> lane q+0, jj>=8 -> lane q+16.
        float lsum = 0.f;
        BF16x16 pb[2];
#pragma unroll
        for (int g = 0; g < 2; ++g) {
            float p0[8], p1[8];
#pragma unroll
            for (int v = 0; v < 8; ++v) {
                p0[v] = exp2_raw(sc[2 * g][v] - m_new);
                p1[v] = exp2_raw(sc[2 * g + 1][v] - m_new);
                lsum += p0[v] + p1[v];
            }
#pragma unroll
            for (int i = 0; i < 4; ++i) {
                unsigned pk0 = pack_bf16(p0[2 * i], p0[2 * i + 1]);
                unsigned pk1 = pack_bf16(p1[2 * i], p1[2 * i + 1]);
                unsigned sw0 = xor16_u32(pk0);
                unsigned sw1 = xor16_u32(pk1);
                pb[g].u[i]     = half ? sw1 : pk0;
                pb[g].u[4 + i] = half ? pk1 : sw0;
            }
        }
        lsum += xor16_f32(lsum);
        s_run = s_run * corr + lsum;
        m_run = m_new;

        // accumulators are (d, q=l16): rescale with OWN lane's corr
#pragma unroll
        for (int db = 0; db < 4; ++db)
#pragma unroll
            for (int v = 0; v < 8; ++v) acc[db][v] *= corr;

        // --- context^T += V^T(16x32) * P^T(32x16), 2 kk-blocks x 4 d-blocks
#pragma unroll
        for (int db = 0; db < 4; ++db) {
            const __bf16* vrow = Vb + (size_t)(db * 16 + l16) * SS + k0;
#pragma unroll
            for (int g = 0; g < 2; ++g) {
                BF16x16 vf;                       // V^T rows as A-matrix
                vf.h[0] = *(const bf16x8*)(vrow + g * 32 + half * 16);
                vf.h[1] = *(const bf16x8*)(vrow + g * 32 + 8 + half * 16);
                acc[db] = wmma_bf16(vf.v, pb[g].v, acc[db]);
            }
        }
    }

    // normalize with OWN lane's sum; pack 8 consecutive d per 16B store
    const float sinv = __builtin_amdgcn_rcpf(s_run);
    const int b = bh / HEADS, h = bh % HEADS;
    const int q = qbase + l16;
    __bf16* orow = Oa + (((size_t)b * SS + q) * HEADS + h) * HDIM;
#pragma unroll
    for (int db = 0; db < 4; ++db) {
        bf16x8 o;
#pragma unroll
        for (int v = 0; v < 8; ++v) o[v] = (__bf16)(acc[db][v] * sinv);
        *(bf16x8*)(orow + db * 16 + half * 8) = o;   // d = db*16 + 8h + v
    }
}

// ---------------------------------------------------------------------------
extern "C" void kernel_launch(void* const* d_in, const int* in_sizes, int n_in,
                              void* d_out, int out_size, void* d_ws, size_t ws_size,
                              hipStream_t stream) {
    const float* q  = (const float*)d_in[0];
    const float* k  = (const float*)d_in[1];
    const float* v  = (const float*)d_in[2];
    const float* wq = (const float*)d_in[3];
    const float* bq = (const float*)d_in[4];
    const float* wk = (const float*)d_in[5];
    const float* bk = (const float*)d_in[6];
    const float* wv = (const float*)d_in[7];
    const float* bv = (const float*)d_in[8];
    const float* wf = (const float*)d_in[9];
    const float* bf = (const float*)d_in[10];

    const size_t ACT = (size_t)MTOK * HID;   // 8,388,608 elems
    const size_t WEL = (size_t)HID * HID;    // 1,048,576 elems

    __bf16* w  = (__bf16*)d_ws;              // ~126 MB bf16 scratch
    __bf16* Xq = w;            __bf16* Xk = Xq + ACT;  __bf16* Xv = Xk + ACT;
    __bf16* Wq = Xv + ACT;     __bf16* Wk = Wq + WEL;
    __bf16* Wv = Wk + WEL;     __bf16* Wf = Wv + WEL;
    __bf16* Qp = Wf + WEL;     __bf16* Kp = Qp + ACT;
    __bf16* Vt = Kp + ACT;     __bf16* Xo = Vt + ACT;

    // 1) fp32 -> bf16
    const int T = 256;
    cvt_f32_bf16<<<(ACT / 4 + T - 1) / T, T, 0, stream>>>(q,  Xq, (int)(ACT / 4));
    cvt_f32_bf16<<<(ACT / 4 + T - 1) / T, T, 0, stream>>>(k,  Xk, (int)(ACT / 4));
    cvt_f32_bf16<<<(ACT / 4 + T - 1) / T, T, 0, stream>>>(v,  Xv, (int)(ACT / 4));
    cvt_f32_bf16<<<(WEL / 4 + T - 1) / T, T, 0, stream>>>(wq, Wq, (int)(WEL / 4));
    cvt_f32_bf16<<<(WEL / 4 + T - 1) / T, T, 0, stream>>>(wk, Wk, (int)(WEL / 4));
    cvt_f32_bf16<<<(WEL / 4 + T - 1) / T, T, 0, stream>>>(wv, Wv, (int)(WEL / 4));
    cvt_f32_bf16<<<(WEL / 4 + T - 1) / T, T, 0, stream>>>(wf, Wf, (int)(WEL / 4));

    // 2) Q/K/V projections (bf16 WMMA, f32 accumulate).
    //    Q gets softmax scale log2(e)/sqrt(64) folded into its epilogue.
    const float SCALE_Q = 0.125f * 1.44269504088896340736f;
    dim3 ggrid(MTOK / 64, HID / 64);         // (128, 16)
    gemm_bf16<0><<<ggrid, 128, 0, stream>>>(Xq, Wq, bq, Qp, SCALE_Q);
    gemm_bf16<0><<<ggrid, 128, 0, stream>>>(Xk, Wk, bk, Kp, 1.0f);
    gemm_bf16<1><<<ggrid, 128, 0, stream>>>(Xv, Wv, bv, Vt, 1.0f);

    // 3) flash attention (DS-free, WMMA + permlanex16 only)
    dim3 agrid(SS / 64, BB * HEADS);         // (32, 64)
    flash_attn<<<agrid, 128, 0, stream>>>(Qp, Kp, Vt, Xo);

    // 4) output projection -> fp32 d_out
    gemm_bf16<2><<<ggrid, 128, 0, stream>>>(Xo, Wf, bf, d_out, 1.0f);
}
